// SimpleMoE_22471268892967
// MI455X (gfx1250) — matrix operand order repaired and checked by reading, not compiled, not run on hardware
//
#include <hip/hip_runtime.h>
#include <hip/hip_bf16.h>
#include <cstddef>
#include <cstdint>

// Problem constants (must match reference)
constexpr int TT   = 8192;   // B*S tokens
constexpr int HH   = 512;    // hidden
constexpr int EE   = 8;      // experts
constexpr int DFFC = 2048;   // ffn dim
constexpr int MT   = 64;     // token tile per workgroup
constexpr int DC   = 64;     // DFF chunk
constexpr int NCH  = EE * (DFFC / DC);  // 256 chunks total (expert-major)

// LDS strides (bf16 elements), padded; all rows 16B aligned
constexpr int XS_STR  = 520; // X tile   [64][520]
constexpr int W1S_STR = 72;  // W1 chunk [512][72] (k-major, n contiguous)
constexpr int AS_STR  = 72;  // act chunk[64][72]
constexpr int W2S_STR = 520; // W2 chunk [64][520] (k-major, c contiguous)

constexpr unsigned XS_OFF   = 0;
constexpr unsigned W1S_OFF  = XS_OFF + MT * XS_STR * 2;     // 66560
constexpr unsigned W2S_OFF0 = W1S_OFF + HH * W1S_STR * 2;   // 140288
constexpr unsigned W2S_OFF1 = W2S_OFF0 + DC * W2S_STR * 2;  // 206848
constexpr unsigned AS_OFF   = W2S_OFF1 + DC * W2S_STR * 2;  // 273408
constexpr unsigned CB_OFF   = AS_OFF + MT * AS_STR * 2;     // 282624
constexpr int SMEM_BYTES    = CB_OFF + MT * EE * 4;         // 284672

typedef __attribute__((ext_vector_type(16))) __bf16 v16bf;
typedef __attribute__((ext_vector_type(8)))  __bf16 v8bf;
typedef __attribute__((ext_vector_type(4)))  __bf16 v4bf;
typedef __attribute__((ext_vector_type(8)))  float  v8f;
typedef __attribute__((ext_vector_type(4)))  float  v4f;

// float4 -> bf16x4 packed convert + 8B LDS store
__device__ inline void st4_bf16(__bf16* dst, v4f v) {
  v4bf o;
#pragma unroll
  for (int i = 0; i < 4; ++i) o[i] = (__bf16)v[i];
  *(v4bf*)dst = o;
}

// ---- CDNA5 async global->LDS copy (16B per lane), tracked by ASYNCcnt ------
__device__ inline void async_cp16(unsigned lds_off, uint64_t gbase, unsigned goff) {
  asm volatile("global_load_async_to_lds_b128 %0, %1, %2"
               :: "v"(lds_off), "v"(goff), "s"(gbase) : "memory");
}
__device__ inline void wait_async16() {
  asm volatile("s_wait_asynccnt 0x10" ::: "memory");
}
__device__ inline void wait_async0() {
  asm volatile("s_wait_asynccnt 0x0" ::: "memory");
}

// ---- WMMA fragment loaders (wave32, 16x16x32 bf16) -------------------------
// A (16x32, MxK): lanes 0-15 -> M=lane, elems0-7:K=0..7, elems8-15:K=16..23;
//                 lanes16-31 -> M=lane-16, K offset +8.
__device__ inline v16bf ld_frag_a(const __bf16* base, int stride, int row0, int k0) {
  const int lane = threadIdx.x & 31;
  const int half = lane >> 4;
  const __bf16* p = base + (row0 + (lane & 15)) * stride + k0 + half * 8;
  v8bf lo = *(const v8bf*)(p);
  v8bf hi = *(const v8bf*)(p + 16);
  v16bf o;
#pragma unroll
  for (int i = 0; i < 8; ++i) { o[i] = lo[i]; o[i + 8] = hi[i]; }
  return o;
}

// B (32x16, KxN): lanes 0-15 -> N=lane, elems0-7:K=0..7, elems8-15:K=16..23;
//                 lanes16-31 -> N=lane-16, K offset +8 (k-major storage, strided gather).
__device__ inline v16bf ld_frag_b(const __bf16* base, int stride, int k0, int n0) {
  const int lane = threadIdx.x & 31;
  const int half = lane >> 4;
  const __bf16* p = base + (k0 + half * 8) * stride + n0 + (lane & 15);
  v16bf o;
#pragma unroll
  for (int i = 0; i < 8; ++i) { o[i] = p[i * stride]; o[i + 8] = p[(i + 16) * stride]; }
  return o;
}

__device__ inline v8f wmma_bf16(v16bf a, v16bf b, v8f c) {
  return __builtin_amdgcn_wmma_f32_16x16x32_bf16(false, a, false, b,
                                                 (short)0, c, false, false);
}

// ---- Kernel: f32 -> bf16 bulk convert (weights, once per launch) ----------
__global__ __launch_bounds__(256)
void cvt_bf16_kernel(const float* __restrict__ src, __bf16* __restrict__ dst,
                     int n4) {
  for (int i = blockIdx.x * 256 + threadIdx.x; i < n4; i += gridDim.x * 256) {
    v4f v = *(const v4f*)(src + (size_t)i * 4);
    v4bf o;
#pragma unroll
    for (int j = 0; j < 4; ++j) o[j] = (__bf16)v[j];
    *(v4bf*)(dst + (size_t)i * 4) = o;
  }
}

// ---- Kernel: zero small accumulators --------------------------------------
__global__ void moe_init_kernel(float* wsf) {
  if (threadIdx.x < 16) wsf[threadIdx.x] = 0.0f;
}

// ---- Kernel: gating (one wave per token) ----------------------------------
__global__ __launch_bounds__(256)
void moe_gate_kernel(const float* __restrict__ x, const float* __restrict__ gW,
                     const float* __restrict__ gb, float* __restrict__ combine,
                     float* __restrict__ prob_acc, float* __restrict__ cnt_acc) {
  const int lane   = threadIdx.x & 31;
  const int gwave  = (blockIdx.x * blockDim.x + threadIdx.x) >> 5;
  const int nwaves = (gridDim.x * blockDim.x) >> 5;

  for (int t = gwave; t < TT; t += nwaves) {
    float a[EE];
#pragma unroll
    for (int e = 0; e < EE; ++e) a[e] = 0.0f;
    for (int h = lane; h < HH; h += 32) {
      float xv = x[(size_t)t * HH + h];
#pragma unroll
      for (int e = 0; e < EE; ++e) a[e] += xv * gW[h * EE + e];
    }
#pragma unroll
    for (int e = 0; e < EE; ++e) {
      float v = a[e];
      for (int off = 16; off > 0; off >>= 1) v += __shfl_down(v, off, 32);
      a[e] = v;
    }
    if (lane == 0) {
      float lg[EE];
#pragma unroll
      for (int e = 0; e < EE; ++e) lg[e] = a[e] + gb[e];
      float mx = lg[0];
#pragma unroll
      for (int e = 1; e < EE; ++e) mx = fmaxf(mx, lg[e]);
      float s = 0.0f, p[EE];
#pragma unroll
      for (int e = 0; e < EE; ++e) { p[e] = expf(lg[e] - mx); s += p[e]; }
      float inv = 1.0f / s;
#pragma unroll
      for (int e = 0; e < EE; ++e) atomicAdd(&prob_acc[e], p[e] * inv);
      int i1 = 0;
#pragma unroll
      for (int e = 1; e < EE; ++e) if (lg[e] > lg[i1]) i1 = e;
      int i2 = (i1 == 0) ? 1 : 0;
#pragma unroll
      for (int e = 0; e < EE; ++e) if (e != i1 && lg[e] > lg[i2]) i2 = e;
      float d  = expf(lg[i2] - lg[i1]);
      float w1 = 1.0f / (1.0f + d);
      float w2 = d * w1;
      float row[EE];
#pragma unroll
      for (int e = 0; e < EE; ++e) row[e] = 0.0f;
      row[i1] = w1; row[i2] += w2;
#pragma unroll
      for (int e = 0; e < EE; ++e) combine[(size_t)t * EE + e] = row[e];
      atomicAdd(&cnt_acc[i1], 1.0f);
      atomicAdd(&cnt_acc[i2], 1.0f);
    }
  }
}

// ---- Async stage helpers: one chunk = DC columns of W1 / DC rows of W2 -----
// Each wave issues exactly 16 async b128 instructions per call.
__device__ inline void issue_w1(int c, int tid, const __bf16* W1b) {
  const int e = c >> 5, dff0 = (c & 31) * DC;
  const uint64_t base = (uint64_t)(W1b + (size_t)e * HH * DFFC);
#pragma unroll
  for (int it = 0; it < 16; ++it) {
    int idx = tid + it * 256;            // 4096 x 16B = 512 rows x 128B
    int k = idx >> 3, seg = idx & 7;
    async_cp16(W1S_OFF + (unsigned)(k * (W1S_STR * 2) + seg * 16), base,
               (unsigned)(((k * DFFC) + dff0 + seg * 8) * 2));
  }
}
__device__ inline void issue_w2(int c, int tid, const __bf16* W2b) {
  const int e = c >> 5, dff0 = (c & 31) * DC;
  const uint64_t base = (uint64_t)(W2b + (size_t)e * DFFC * HH);
  const unsigned lds0 = (c & 1) ? W2S_OFF1 : W2S_OFF0;
#pragma unroll
  for (int it = 0; it < 16; ++it) {
    int idx = tid + it * 256;            // 4096 x 16B = 64 rows x 1024B
    int k = idx >> 6, seg = idx & 63;
    async_cp16(lds0 + (unsigned)(k * (W2S_STR * 2) + seg * 16), base,
               (unsigned)((((dff0 + k) * HH) + seg * 8) * 2));
  }
}

// ---- Kernel: fused dense MoE FFN (WMMA bf16, fp32 accumulate) -------------
__global__ __launch_bounds__(256, 1)
void moe_ffn_kernel(const float* __restrict__ x,
                    const __bf16* __restrict__ W1b, const float* __restrict__ b1,
                    const __bf16* __restrict__ W2b, const float* __restrict__ b2,
                    const float* __restrict__ combine, float* __restrict__ out) {
  extern __shared__ __align__(16) char smem[];
  __bf16* Xs    = (__bf16*)(smem + XS_OFF);
  __bf16* W1s   = (__bf16*)(smem + W1S_OFF);
  __bf16* As    = (__bf16*)(smem + AS_OFF);
  float*  combs = (float*)(smem + CB_OFF);

  const int tid  = threadIdx.x;
  const int wave = tid >> 5;
  const int half = (tid >> 4) & 1;
  const int l15  = tid & 15;
  const int t0   = blockIdx.x * MT;
  const int mwA  = wave >> 2;   // GEMM1: m-tiles mwA and mwA+2
  const int nw   = wave & 3;    // GEMM1: n-tile within DFF chunk
  const int c0   = wave * 64;   // GEMM2: output column slice

  // Stage X tile (f32 -> bf16), vectorized float4
  const float* xg = x + (size_t)t0 * HH;
  for (int i = tid; i < MT * HH / 4; i += 256) {
    int r = i >> 7, c4 = (i & 127) << 2;
    st4_bf16(&Xs[r * XS_STR + c4], *(const v4f*)(xg + r * HH + c4));
  }
#pragma unroll
  for (int i = tid; i < MT * EE; i += 256)
    combs[i] = combine[(size_t)(t0 + (i >> 3)) * EE + (i & 7)];

  v8f acc[4][4];
#pragma unroll
  for (int mt = 0; mt < 4; ++mt)
#pragma unroll
    for (int nt = 0; nt < 4; ++nt)
#pragma unroll
      for (int j = 0; j < 8; ++j) acc[mt][nt][j] = 0.0f;

  // Pipeline prologue: chunk 0 in flight
  issue_w1(0, tid, W1b);
  issue_w2(0, tid, W2b);

  for (int c = 0; c < NCH; ++c) {
    // 1) issue next W2 into the ping-pong buffer just freed by GEMM2(c-1)
    if (c + 1 < NCH) {
      issue_w2(c + 1, tid, W2b);
      wait_async16();                 // drain W1(c), W2(c); W2(c+1) in flight
    } else {
      wait_async0();
    }
    __syncthreads();                  // chunk c weights visible to all waves

    const int e = c >> 5;
    const int dff0 = (c & 31) * DC;
    const __bf16* W2cur = (const __bf16*)(smem + ((c & 1) ? W2S_OFF1 : W2S_OFF0));

    // GEMM1: two 16x16 tiles of gelu(X @ W1_chunk + b1) per wave
    v8f c1[2];
#pragma unroll
    for (int u = 0; u < 2; ++u)
#pragma unroll
      for (int j = 0; j < 8; ++j) c1[u][j] = 0.0f;
    for (int k0 = 0; k0 < HH; k0 += 32) {
      v16bf b  = ld_frag_b(W1s, W1S_STR, k0, nw * 16);
      v16bf a0 = ld_frag_a(Xs, XS_STR, mwA * 16, k0);
      v16bf a1 = ld_frag_a(Xs, XS_STR, (mwA + 2) * 16, k0);
      c1[0] = wmma_bf16(a0, b, c1[0]);
      c1[1] = wmma_bf16(a1, b, c1[1]);
    }
    const int nl = nw * 16 + l15;
    const float bias = b1[(size_t)e * DFFC + dff0 + nl];
#pragma unroll
    for (int u = 0; u < 2; ++u) {
      const int mt = mwA + 2 * u;
#pragma unroll
      for (int j = 0; j < 8; ++j) {
        int m = mt * 16 + half * 8 + j;          // C layout: M = j + 8*half
        float v = c1[u][j] + bias;
        float g = 0.5f * v * (1.0f + erff(v * 0.70710678118654752f));
        As[m * AS_STR + nl] = (__bf16)(g * combs[m * EE + e]);  // fold combine
      }
    }
    __syncthreads();                  // As ready; W1s fully consumed
    // 2) refill the single W1 buffer for chunk c+1 (overlaps with GEMM2)
    if (c + 1 < NCH) issue_w1(c + 1, tid, W1b);

    // GEMM2: out_tile[:, c0:c0+64] += As @ W2_chunk
#pragma unroll
    for (int kk = 0; kk < DC; kk += 32) {
      v16bf bw[4];
#pragma unroll
      for (int nt = 0; nt < 4; ++nt)
        bw[nt] = ld_frag_b(W2cur, W2S_STR, kk, c0 + nt * 16);
#pragma unroll
      for (int mt = 0; mt < 4; ++mt) {
        v16bf a = ld_frag_a(As, AS_STR, mt * 16, kk);
#pragma unroll
        for (int nt = 0; nt < 4; ++nt)
          acc[mt][nt] = wmma_bf16(a, bw[nt], acc[mt][nt]);
      }
    }
    __syncthreads();                  // W2buf[c&1] + As free for reuse
  }

  // Epilogue: add sum_e combine[t,e]*b2[e][c] and store
#pragma unroll
  for (int nt = 0; nt < 4; ++nt) {
    int c = c0 + nt * 16 + l15;
    float bv[EE];
#pragma unroll
    for (int e = 0; e < EE; ++e) bv[e] = b2[e * HH + c];
#pragma unroll
    for (int mt = 0; mt < 4; ++mt)
#pragma unroll
      for (int j = 0; j < 8; ++j) {
        int m = mt * 16 + half * 8 + j;
        float bs = 0.0f;
#pragma unroll
        for (int e = 0; e < EE; ++e) bs += combs[m * EE + e] * bv[e];
        out[(size_t)(t0 + m) * HH + c] = acc[mt][nt][j] + bs;
      }
  }
}

// ---- Kernel: finalize l_aux + expert counts -------------------------------
__global__ void moe_finalize_kernel(const float* __restrict__ prob_acc,
                                    const float* __restrict__ cnt_acc,
                                    float* __restrict__ tail) {
  int t = threadIdx.x;
  if (t == 0) {
    float s = 0.0f;
#pragma unroll
    for (int e = 0; e < EE; ++e) {
      float pe = prob_acc[e] * (1.0f / (float)TT);
      s += pe * pe;
    }
    tail[0] = s * (float)EE;
  }
  if (t < EE) tail[1 + t] = cnt_acc[t];
}

// ---- Host launch -----------------------------------------------------------
extern "C" void kernel_launch(void* const* d_in, const int* in_sizes, int n_in,
                              void* d_out, int out_size, void* d_ws, size_t ws_size,
                              hipStream_t stream) {
  const float* x  = (const float*)d_in[0];
  const float* gW = (const float*)d_in[1];
  const float* gb = (const float*)d_in[2];
  const float* W1 = (const float*)d_in[3];
  const float* b1 = (const float*)d_in[4];
  const float* W2 = (const float*)d_in[5];
  const float* b2 = (const float*)d_in[6];
  float* out = (float*)d_out;

  // Workspace layout (bytes):
  //   [0,64)        prob_acc[8] + cnt_acc[8]
  //   [256, 262400) combine [TT*EE] f32
  //   [262400, +16.78MB) W1 bf16 ; then W2 bf16
  float*  wsf      = (float*)d_ws;
  float*  prob_acc = wsf;
  float*  cnt_acc  = wsf + 8;
  float*  combine  = (float*)((char*)d_ws + 256);
  __bf16* W1bf     = (__bf16*)((char*)d_ws + 262400);
  __bf16* W2bf     = W1bf + (size_t)EE * HH * DFFC;

  (void)hipFuncSetAttribute((const void*)moe_ffn_kernel,
                            hipFuncAttributeMaxDynamicSharedMemorySize,
                            SMEM_BYTES);

  const int n4 = EE * HH * DFFC / 4;  // 2,097,152 float4 per weight tensor
  moe_init_kernel<<<1, 32, 0, stream>>>(wsf);
  cvt_bf16_kernel<<<4096, 256, 0, stream>>>(W1, W1bf, n4);
  cvt_bf16_kernel<<<4096, 256, 0, stream>>>(W2, W2bf, n4);
  moe_gate_kernel<<<256, 256, 0, stream>>>(x, gW, gb, combine, prob_acc, cnt_acc);
  moe_ffn_kernel<<<TT / MT, 256, SMEM_BYTES, stream>>>(x, W1bf, b1, W2bf, b2,
                                                       combine, out);
  moe_finalize_kernel<<<1, 32, 0, stream>>>(prob_acc, cnt_acc,
                                            out + (size_t)TT * HH);
}
